// NNUENet_71356586655948
// MI455X (gfx1250) — matrix operand order repaired and checked by reading, not compiled, not run on hardware
//
#include <hip/hip_runtime.h>

// ---------------- problem constants (from reference) ----------------
#define FEATURE_SIZE 41024
#define BATCH        8192
#define MAX_ACTIVE   32
#define H1_DIM       256
#define H2_DIM       32

typedef __attribute__((ext_vector_type(16))) _Float16 v16h;
typedef __attribute__((ext_vector_type(8)))  float    v8f;

// workspace layout
//   [0, 21004288)            : wT  = fc1_w transposed, f16  [41024][256]
//   [21004288, 21004288+16K) : bp  = fc2_w pre-packed B fragments, f16
#define WT_BYTES ((size_t)FEATURE_SIZE * H1_DIM * 2)

// --------------------------------------------------------------------
// Kernel A: transpose fc1_w (256 x 41024, f32 row-major) into
// wT (41024 x 256, f16 row-major): each feature's slice becomes one
// contiguous 512B row -> coalesced L2-resident gathers in the forward.
// --------------------------------------------------------------------
__global__ __launch_bounds__(256)
void nnue_transpose_w1(const float* __restrict__ W, _Float16* __restrict__ WT) {
    __shared__ float tile[32][33];          // +1 pad: avoid bank conflicts
    const int f0 = blockIdx.x * 32;         // feature tile base (gridDim.x = 1282)
    const int h0 = blockIdx.y * 32;         // hidden tile base  (gridDim.y = 8)
    const int tx = threadIdx.x;             // 0..31
    const int ty = threadIdx.y;             // 0..7

    #pragma unroll
    for (int i = 0; i < 32; i += 8) {
        const int h = h0 + ty + i;
        tile[ty + i][tx] = W[(size_t)h * FEATURE_SIZE + f0 + tx];
    }
    __syncthreads();
    #pragma unroll
    for (int i = 0; i < 32; i += 8) {
        const int f = f0 + ty + i;
        WT[(size_t)f * H1_DIM + h0 + tx] = (_Float16)tile[tx][ty + i];
    }
}

// --------------------------------------------------------------------
// Kernel A2: pre-pack fc2_w into f16 WMMA B-fragments.
// Fragment layout: bp[ntile][kc][lane][e]  (ntile 0..1, kc 0..7,
// lane 0..31, e 0..15) where, in this code's contiguous k order (A matches),
//   n = ntile*16 + (lane & 15),  k = kc*32 + (lane>>4)*16 + e.
// Each WMMA lane then loads its whole fragment as one contiguous v16h.
// --------------------------------------------------------------------
__global__ __launch_bounds__(256)
void nnue_pack_w2(const float* __restrict__ w2, _Float16* __restrict__ bp) {
    const int t = threadIdx.x;                  // 2*8*32*16 = 8192 halves
    #pragma unroll
    for (int i = t; i < 8192; i += 256) {
        const int e    =  i        & 15;
        const int lane = (i >> 4)  & 31;
        const int kc   = (i >> 9)  & 7;
        const int nt   =  i >> 12;
        const int n = nt * 16 + (lane & 15);
        const int k = kc * 32 + (lane >> 4) * 16 + e;
        bp[i] = (_Float16)w2[n * H1_DIM + k];
    }
}

// --------------------------------------------------------------------
// Kernel B: fused forward. One block = 16 batch rows, 256 threads (8 waves).
//   Stage 1: sparse gather-sum -> h1 (f16 in LDS, clipped to [0,1])
//   Stage 2: fc2 via v_wmma_f32_16x16x32_f16 (waves 0,1; K=256 -> 8 WMMAs)
//   Stage 3: fc3 dot(32) + tanh (16 lanes)
// --------------------------------------------------------------------
#define H1_PAD (H1_DIM + 8)   // LDS row stride in halves; offsets banks

__global__ __launch_bounds__(256)
void nnue_forward(const int*      __restrict__ active_idx,  // [BATCH][32]
                  const _Float16* __restrict__ wT,          // [F][256] f16
                  const v16h*     __restrict__ bp,          // packed B frags
                  const float*    __restrict__ b1,          // [256]
                  const float*    __restrict__ b2,          // [32]
                  const float*    __restrict__ w3,          // [32]
                  const float*    __restrict__ b3,          // [1]
                  float*          __restrict__ out)         // [BATCH]
{
    __shared__ _Float16 h1s[16][H1_PAD];     // 16 rows x 256 (padded)  ~8.4 KB
    __shared__ float    h2s[16][H2_DIM];     // 16 rows x 32             2 KB
    __shared__ int      idxs[16 * MAX_ACTIVE];

    const int t  = threadIdx.x;              // 0..255
    const int b0 = blockIdx.x * 16;          // batch tile base (512 blocks)

    // ---- stage the 512 indices of this batch tile into LDS; prefetch
    //      the corresponding wT rows into cache (global_prefetch_b8) ----
    {
        const int i0 = active_idx[b0 * MAX_ACTIVE + t];
        const int i1 = active_idx[b0 * MAX_ACTIVE + t + 256];
        idxs[t]       = i0;
        idxs[t + 256] = i1;
        if (i0 >= 0) __builtin_prefetch(&wT[(size_t)i0 * H1_DIM], 0, 0);
        if (i1 >= 0) __builtin_prefetch(&wT[(size_t)i1 * H1_DIM], 0, 0);
    }
    __syncthreads();

    // ---- Stage 1: gather-sum fc1. Thread t owns hidden column t. ----
    // All 256 lanes of the block read consecutive halves of one 512B
    // feature row per (r, j) step -> fully coalesced L2-resident loads.
    const float bias1 = b1[t];
    #pragma unroll
    for (int r = 0; r < 16; ++r) {
        float acc = bias1;
        #pragma unroll
        for (int j = 0; j < MAX_ACTIVE; ++j) {
            const int ix = idxs[r * MAX_ACTIVE + j];    // uniform per step
            if (ix >= 0)
                acc += (float)wT[(size_t)ix * H1_DIM + t];
        }
        acc = fminf(fmaxf(acc, 0.0f), 1.0f);            // clip [0,1]
        h1s[r][t] = (_Float16)acc;
    }
    __syncthreads();

    // ---- Stage 2: fc2 as WMMA. Waves 0,1 -> output cols 0-15 / 16-31. ----
    if (t < 64) {
        const int lane   = t & 31;
        const int ntile  = t >> 5;             // wave 0: N 0-15, wave 1: N 16-31
        const int m      = lane & 15;          // batch row within tile
        const int n      = ntile * 16 + (lane & 15);
        const int hihalf = lane >> 4;          // lane group 0 / 1

        v8f c = {};
        #pragma unroll
        for (int kc = 0; kc < 8; ++kc) {       // K = 256 = 8 * 32
            const int k0 = kc * 32;
            // A 16x32 f16 layout (matches packed B): lanes0-15 K=0..15 ;
            //                                        lanes16-31 K=16..31
            v16h a = {};
            const int akoff = hihalf * 16;
            #pragma unroll
            for (int e = 0; e < 16; ++e) {
                const int ka = k0 + akoff + e;
                a[e] = h1s[m][ka];
            }
            // B fragment: pre-packed contiguous 32B per lane
            const v16h b = bp[(ntile * 8 + kc) * 32 + lane];
            c = __builtin_amdgcn_wmma_f32_16x16x32_f16(
                    /*neg_a=*/false, a, /*neg_b=*/false, b,
                    /*c_mod=*/(short)0, c, /*reuse_a=*/false, /*reuse_b=*/false);
        }
        // C/D layout: VGPR r -> row (r + 8*hihalf), col = n
        const int rowbase = hihalf * 8;
        const float bias2 = b2[n];
        #pragma unroll
        for (int r = 0; r < 8; ++r) {
            float v = c[r] + bias2;
            v = fminf(fmaxf(v, 0.0f), 1.0f);            // clip [0,1]
            h2s[rowbase + r][n] = v;
        }
    }
    __syncthreads();

    // ---- Stage 3: fc3 (dot over 32) + tanh ----
    if (t < 16) {
        float s = b3[0];
        #pragma unroll
        for (int o = 0; o < H2_DIM; ++o)
            s += h2s[t][o] * w3[o];
        out[b0 + t] = tanhf(s);
    }
}

// --------------------------------------------------------------------
// Launch: transpose fc1_w + pack fc2_w into workspace, then fused forward.
// --------------------------------------------------------------------
extern "C" void kernel_launch(void* const* d_in, const int* in_sizes, int n_in,
                              void* d_out, int out_size, void* d_ws, size_t ws_size,
                              hipStream_t stream) {
    const int*   idx   = (const int*)  d_in[0];   // active_indices [8192][32]
    const float* fc1_w = (const float*)d_in[1];   // [256][41024]
    const float* fc1_b = (const float*)d_in[2];   // [256]
    const float* fc2_w = (const float*)d_in[3];   // [32][256]
    const float* fc2_b = (const float*)d_in[4];   // [32]
    const float* fc3_w = (const float*)d_in[5];   // [1][32]
    const float* fc3_b = (const float*)d_in[6];   // [1]
    float*       out   = (float*)d_out;           // [8192]

    _Float16* wT = (_Float16*)d_ws;                       // 21 MB f16 table
    _Float16* bp = (_Float16*)((char*)d_ws + WT_BYTES);   // 16 KB packed B

    dim3 tb_t(32, 8);
    dim3 tg_t(FEATURE_SIZE / 32, H1_DIM / 32);    // 1282 x 8
    nnue_transpose_w1<<<tg_t, tb_t, 0, stream>>>(fc1_w, wT);
    nnue_pack_w2<<<1, 256, 0, stream>>>(fc2_w, bp);

    nnue_forward<<<BATCH / 16, 256, 0, stream>>>(idx, wT, (const v16h*)bp,
                                                 fc1_b, fc2_b, fc3_w, fc3_b, out);
}
